// ODE_VAE_Decoder_69630009802958
// MI455X (gfx1250) — matrix operand order repaired and checked by reading, not compiled, not run on hardware
//
#include <hip/hip_runtime.h>
#include <hip/hip_bf16.h>
#include <math.h>

// ---------------------------------------------------------------------------
// ODE-VAE decoder for MI455X (gfx1250): bf16 WMMA GEMMs with async-to-LDS
// double buffering (no VGPR staging, no spills), bf16 activation chain.
// ---------------------------------------------------------------------------

typedef __attribute__((ext_vector_type(16))) __bf16 v16bf;
typedef __attribute__((ext_vector_type(8)))  float  v8f;

#define TILE_M 128
#define TILE_N 128
#define TILE_K 64
#define LDS_STR 72   // padded bf16 stride: 144B/row -> conflict-free frag gathers

__device__ __forceinline__ unsigned short f2bf(float f) {
    unsigned u = __float_as_uint(f);
    unsigned r = u + 0x7FFFu + ((u >> 16) & 1u);   // round-to-nearest-even
    return (unsigned short)(r >> 16);
}

// Branch-free tanh: 1 - 2/(exp(2x)+1); saturates correctly at +/-inf.
__device__ __forceinline__ float fast_tanh(float x) {
    float e = __expf(2.0f * x);                    // v_exp_f32
    return 1.0f - 2.0f * __builtin_amdgcn_rcpf(e + 1.0f);
}

// One 16B async memory->LDS copy per lane (ASYNCcnt-tracked).
__device__ __forceinline__ void async_b128(unsigned lds_off, const void* gaddr) {
    unsigned long long g = (unsigned long long)gaddr;
    asm volatile("global_load_async_to_lds_b128 %0, %1, off"
                 :: "v"(lds_off), "v"(g) : "memory");
}
__device__ __forceinline__ void wait_async0() {
    asm volatile("s_wait_asynccnt 0x0" ::: "memory");
}

// Convert W[K][N] (fp32, row-major) -> Wt[N][K] (bf16 bits), coalesced writes.
__global__ void wconvert_t(const float* __restrict__ W, unsigned short* __restrict__ Wt,
                           int K, int N) {
    long idx = (long)blockIdx.x * blockDim.x + threadIdx.x;
    long total = (long)K * N;
    if (idx >= total) return;
    int k = (int)(idx % K);
    int n = (int)(idx / K);
    Wt[idx] = f2bf(W[(long)k * N + n]);
}

__global__ void f32_to_bf16(const float* __restrict__ x, unsigned short* __restrict__ o, long n) {
    long i = (long)blockIdx.x * blockDim.x + threadIdx.x;
    if (i < n) o[i] = f2bf(x[i]);
}

// ---------------------------------------------------------------------------
// C[M,N] = act(Abf[M,K] @ Bt[N,K]^T + bias),  Abf/Bt bf16-bits, acc fp32.
// ACT: 0=identity, 1=tanh.  OUT: 0=f32, 1=bf16, 2=both.
// M%128==0, N%128==0, K%64==0.  Block 256 thr = 8 wave32 (2x4 waves),
// wave tile 64x32 = 4x2 WMMA frags.  Double-buffered LDS filled by
// global_load_async_to_lds_b128 so next-tile copies overlap the WMMAs.
// ---------------------------------------------------------------------------
template <int ACT, int OUT>
__global__ __launch_bounds__(256)
void gemm_bf16_wmma(const unsigned short* __restrict__ Abf,
                    const unsigned short* __restrict__ Bt,
                    const float* __restrict__ bias,
                    float* __restrict__ Cf, unsigned short* __restrict__ Cbf,
                    int M, int N, int K) {
    __shared__ __align__(16) unsigned short ldsA[2][TILE_M * LDS_STR];
    __shared__ __align__(16) unsigned short ldsB[2][TILE_N * LDS_STR];

    const int tid  = threadIdx.x;
    const int lane = tid & 31;
    const int wave = tid >> 5;
    const int wm   = wave & 1;   // 2 waves along M  (64 rows each)
    const int wn   = wave >> 1;  // 4 waves along N  (32 cols each)
    const long blockM = (long)blockIdx.y * TILE_M;
    const long blockN = (long)blockIdx.x * TILE_N;

    v8f acc[4][2];
#pragma unroll
    for (int i = 0; i < 4; ++i)
#pragma unroll
        for (int j = 0; j < 2; ++j)
            acc[i][j] = (v8f){0.f, 0.f, 0.f, 0.f, 0.f, 0.f, 0.f, 0.f};

    const int ml   = lane & 15;
    const int half = lane >> 4;

    // Issue async copies for tile at k-offset `kt` into LDS buffer `buf`.
    // A tile: 128 rows x 64 bf16 = 1024 x 16B ; B tile identical shape.
    auto prefetch = [&](int kt, int buf) {
#pragma unroll
        for (int i = 0; i < 4; ++i) {
            int idx = tid + i * 256;            // 0..1023
            int row = idx >> 3;
            int kq  = idx & 7;
            unsigned loff = (unsigned)(unsigned long long)&ldsA[buf][row * LDS_STR + kq * 8];
            async_b128(loff, Abf + (blockM + row) * (long)K + kt + kq * 8);
        }
#pragma unroll
        for (int i = 0; i < 4; ++i) {
            int idx = tid + i * 256;
            int n  = idx >> 3;
            int kq = idx & 7;
            unsigned loff = (unsigned)(unsigned long long)&ldsB[buf][n * LDS_STR + kq * 8];
            async_b128(loff, Bt + (blockN + n) * (long)K + kt + kq * 8);
        }
    };

    prefetch(0, 0);
    wait_async0();
    __syncthreads();

    const int nk = K / TILE_K;
    for (int it = 0; it < nk; ++it) {
        const int cur = it & 1;
        if (it + 1 < nk) prefetch((it + 1) * TILE_K, cur ^ 1);  // overlaps WMMAs

#pragma unroll
        for (int kc = 0; kc < 2; ++kc) {             // two K=32 WMMA slices
            union Frag { uint4 q[2]; v16bf v; };
            Frag af[4], bf[2];
            // A 16x32 bf16: lanes 0-15 -> K 0-7/16-23, lanes 16-31 -> K 8-15/24-31
#pragma unroll
            for (int fm = 0; fm < 4; ++fm) {
                const unsigned short* p =
                    &ldsA[cur][(wm * 64 + fm * 16 + ml) * LDS_STR + kc * 32 + half * 8];
                af[fm].q[0] = *reinterpret_cast<const uint4*>(p);
                af[fm].q[1] = *reinterpret_cast<const uint4*>(p + 16);
            }
            // B 32x16 bf16: lane = N, lanes 0-15 -> K 0-7/16-23, lanes 16-31 -> K 8-15/24-31
#pragma unroll
            for (int fn = 0; fn < 2; ++fn) {
                const unsigned short* p =
                    &ldsB[cur][(wn * 32 + fn * 16 + ml) * LDS_STR + kc * 32 + half * 8];
                bf[fn].q[0] = *reinterpret_cast<const uint4*>(p);
                bf[fn].q[1] = *reinterpret_cast<const uint4*>(p + 16);
            }
#pragma unroll
            for (int fm = 0; fm < 4; ++fm)
#pragma unroll
                for (int fn = 0; fn < 2; ++fn)
                    acc[fm][fn] = __builtin_amdgcn_wmma_f32_16x16x32_bf16(
                        false, af[fm].v, false, bf[fn].v,
                        (short)0, acc[fm][fn], false, false);
        }

        if (it + 1 < nk) {
            wait_async0();          // next tile resident in LDS
            __syncthreads();
        }
    }

    // ---- epilogue: C/D layout VGPR r -> M = r + 8*half, N = lane&15 ----
#pragma unroll
    for (int fm = 0; fm < 4; ++fm) {
#pragma unroll
        for (int fn = 0; fn < 2; ++fn) {
            long col = blockN + wn * 32 + fn * 16 + ml;
            float bv = bias[col];
#pragma unroll
            for (int r = 0; r < 8; ++r) {
                long row = blockM + wm * 64 + fm * 16 + half * 8 + r;
                float v = acc[fm][fn][r] + bv;
                if (ACT == 1) v = fast_tanh(v);
                if (OUT == 0 || OUT == 2) Cf[row * (long)N + col] = v;
                if (OUT == 1 || OUT == 2) Cbf[row * (long)N + col] = f2bf(v);
            }
        }
    }
}

// r_bf = bf16(mu + exp(0.5*log_var) * eps)
__global__ void reparam_kernel(const float* __restrict__ mu, const float* __restrict__ lv,
                               const float* __restrict__ eps,
                               unsigned short* __restrict__ r_bf, long n) {
    long i = (long)blockIdx.x * blockDim.x + threadIdx.x;
    if (i < n) r_bf[i] = f2bf(mu[i] + __expf(0.5f * lv[i]) * eps[i]);
}

// RK4 stages; dt from device timestamps.
// stages 0-2: ytmp_bf = bf16(y + c*dt*k), acc updated.
// stage  3  : y (f32) and y_bf updated with the combined increment.
__global__ void rk4_stage(const float* __restrict__ y, const float* __restrict__ k,
                          float* yo_f32, unsigned short* __restrict__ yo_bf,
                          float* acc, const float* __restrict__ ts,
                          int ti, int stage, long n) {
    long i = (long)blockIdx.x * blockDim.x + threadIdx.x;
    if (i >= n) return;
    float dt = ts[ti + 1] - ts[ti];
    float kv = k[i];
    float t;
    if (stage == 0)      { t = y[i] + 0.5f * dt * kv; acc[i] = kv; }
    else if (stage == 1) { t = y[i] + 0.5f * dt * kv; acc[i] += 2.f * kv; }
    else if (stage == 2) { t = y[i] +        dt * kv; acc[i] += 2.f * kv; }
    else                 { t = y[i] + (dt / 6.f) * (acc[i] + kv); yo_f32[i] = t; }
    yo_bf[i] = f2bf(t);
}

// Row softmax over 128 classes; one wave32 per row, 4 elements per lane.
__global__ void softmax128(const float* __restrict__ logits, float* __restrict__ out, int rows) {
    int wave = threadIdx.x >> 5;
    int lane = threadIdx.x & 31;
    int row  = blockIdx.x * 8 + wave;
    if (row >= rows) return;
    const float* p = logits + (long)row * 128;
    float v[4];
    float mx = -3.4e38f;
#pragma unroll
    for (int i = 0; i < 4; ++i) { v[i] = p[lane + i * 32]; mx = fmaxf(mx, v[i]); }
#pragma unroll
    for (int s = 16; s > 0; s >>= 1) mx = fmaxf(mx, __shfl_xor(mx, s, 32));
    float sum = 0.f;
#pragma unroll
    for (int i = 0; i < 4; ++i) { v[i] = __expf(v[i] - mx); sum += v[i]; }
#pragma unroll
    for (int s = 16; s > 0; s >>= 1) sum += __shfl_xor(sum, s, 32);
    float inv = 1.f / sum;
    float* o = out + (long)row * 128;
#pragma unroll
    for (int i = 0; i < 4; ++i) o[lane + i * 32] = v[i] * inv;
}

// ---------------------------------------------------------------------------
extern "C" void kernel_launch(void* const* d_in, const int* in_sizes, int n_in,
                              void* d_out, int out_size, void* d_ws, size_t ws_size,
                              hipStream_t stream) {
    (void)in_sizes; (void)n_in; (void)out_size; (void)ws_size;
    enum { Bn = 4096, Hh = 1024, Dd = 256, OHh = 2048, Cc = 128, Tt = 21 };

    const float* h       = (const float*)d_in[0];
    const float* eps     = (const float*)d_in[1];
    const float* ts      = (const float*)d_in[2];
    const float* fc_mu_W = (const float*)d_in[3];
    const float* fc_mu_b = (const float*)d_in[4];
    const float* fc_var_W= (const float*)d_in[5];
    const float* fc_var_b= (const float*)d_in[6];
    const float* map_W   = (const float*)d_in[7];
    const float* map_b   = (const float*)d_in[8];
    const float* ode_W1  = (const float*)d_in[9];
    const float* ode_b1  = (const float*)d_in[10];
    const float* ode_W2  = (const float*)d_in[11];
    const float* ode_b2  = (const float*)d_in[12];
    const float* fin_W   = (const float*)d_in[13];
    const float* fin_b   = (const float*)d_in[14];

    float* out    = (float*)d_out;
    float* out_sm = out;                                  // [T,B,C]
    float* out_mu = out + (long)Tt * Bn * Cc;             // [B,D]
    float* out_lv = out_mu + (long)Bn * Dd;               // [B,D]

    // ---- workspace carve-up (256B aligned) ----
    char* w = (char*)d_ws;
    auto carve = [&](size_t bytes) -> char* {
        char* p = w; w += (bytes + 255) & ~(size_t)255; return p;
    };
    unsigned short* wT_mu   = (unsigned short*)carve((size_t)Hh * Dd * 2);
    unsigned short* wT_var  = (unsigned short*)carve((size_t)Hh * Dd * 2);
    unsigned short* wT_map  = (unsigned short*)carve((size_t)Dd * Hh * 2);
    unsigned short* wT_1    = (unsigned short*)carve((size_t)Hh * OHh * 2);
    unsigned short* wT_2    = (unsigned short*)carve((size_t)OHh * Hh * 2);
    unsigned short* wT_fin  = (unsigned short*)carve((size_t)Hh * Cc * 2);
    unsigned short* h_bf    = (unsigned short*)carve((size_t)Bn * Hh * 2);
    unsigned short* r_bf    = (unsigned short*)carve((size_t)Bn * Dd * 2);
    unsigned short* y_bf    = (unsigned short*)carve((size_t)Bn * Hh * 2);
    unsigned short* ytmp_bf = (unsigned short*)carve((size_t)Bn * Hh * 2);
    unsigned short* hid_bf  = (unsigned short*)carve((size_t)Bn * OHh * 2);
    float* y      = (float*)carve((size_t)Bn * Hh * 4);
    float* acc    = (float*)carve((size_t)Bn * Hh * 4);
    float* kbuf   = (float*)carve((size_t)Bn * Hh * 4);
    float* logits = (float*)carve((size_t)Bn * Cc * 4);

    // ---- convert weights (transposed to [N][K]) and h to bf16 ----
    auto conv = [&](const float* W, unsigned short* Wt, int K, int N) {
        long total = (long)K * N;
        wconvert_t<<<dim3((unsigned)((total + 255) / 256)), dim3(256), 0, stream>>>(W, Wt, K, N);
    };
    conv(fc_mu_W,  wT_mu,  Hh,  Dd);
    conv(fc_var_W, wT_var, Hh,  Dd);
    conv(map_W,    wT_map, Dd,  Hh);
    conv(ode_W1,   wT_1,   Hh,  OHh);
    conv(ode_W2,   wT_2,   OHh, Hh);
    conv(fin_W,    wT_fin, Hh,  Cc);
    {
        long n = (long)Bn * Hh;
        f32_to_bf16<<<dim3((unsigned)((n + 255) / 256)), dim3(256), 0, stream>>>(h, h_bf, n);
    }

    auto gemm = [&](int act, int outm,
                    const unsigned short* A, const unsigned short* Bt, const float* bias,
                    float* Cf, unsigned short* Cbf, int M, int N, int K) {
        dim3 grid((unsigned)(N / TILE_N), (unsigned)(M / TILE_M));
        if (act == 1)
            gemm_bf16_wmma<1, 1><<<grid, dim3(256), 0, stream>>>(A, Bt, bias, Cf, Cbf, M, N, K);
        else if (outm == 2)
            gemm_bf16_wmma<0, 2><<<grid, dim3(256), 0, stream>>>(A, Bt, bias, Cf, Cbf, M, N, K);
        else
            gemm_bf16_wmma<0, 0><<<grid, dim3(256), 0, stream>>>(A, Bt, bias, Cf, Cbf, M, N, K);
    };

    const long nBH = (long)Bn * Hh;
    const unsigned gBH = (unsigned)((nBH + 255) / 256);
    auto ode_f = [&](const unsigned short* yin_bf) {   // kbuf = f(yin)
        gemm(1, 1, yin_bf, wT_1, ode_b1, nullptr, hid_bf, Bn, OHh, Hh);   // tanh, bf16 out
        gemm(0, 0, hid_bf, wT_2, ode_b2, kbuf,    nullptr, Bn, Hh,  OHh); // f32 out
    };
    auto project = [&](int t) {                        // out[t] = softmax(y @ finW + b)
        gemm(0, 0, y_bf, wT_fin, fin_b, logits, nullptr, Bn, Cc, Hh);
        softmax128<<<dim3(Bn / 8), dim3(256), 0, stream>>>(logits, out_sm + (long)t * Bn * Cc, Bn);
    };

    // ---- encoder head: mu, log_var, reparameterize, map back ----
    gemm(0, 0, h_bf, wT_mu,  fc_mu_b,  out_mu, nullptr, Bn, Dd, Hh);
    gemm(0, 0, h_bf, wT_var, fc_var_b, out_lv, nullptr, Bn, Dd, Hh);
    {
        long n = (long)Bn * Dd;
        reparam_kernel<<<dim3((unsigned)((n + 255) / 256)), dim3(256), 0, stream>>>(
            out_mu, out_lv, eps, r_bf, n);
    }
    gemm(0, 2, r_bf, wT_map, map_b, y, y_bf, Bn, Hh, Dd);   // z -> y (f32 + bf16)

    project(0);

    // ---- RK4 time stepping ----
    for (int t = 1; t < Tt; ++t) {
        int ti = t - 1;
        ode_f(y_bf);                                                          // k1
        rk4_stage<<<gBH, 256, 0, stream>>>(y, kbuf, nullptr, ytmp_bf, acc, ts, ti, 0, nBH);
        ode_f(ytmp_bf);                                                       // k2
        rk4_stage<<<gBH, 256, 0, stream>>>(y, kbuf, nullptr, ytmp_bf, acc, ts, ti, 1, nBH);
        ode_f(ytmp_bf);                                                       // k3
        rk4_stage<<<gBH, 256, 0, stream>>>(y, kbuf, nullptr, ytmp_bf, acc, ts, ti, 2, nBH);
        ode_f(ytmp_bf);                                                       // k4
        rk4_stage<<<gBH, 256, 0, stream>>>(y, kbuf, y, y_bf, acc, ts, ti, 3, nBH);
        project(t);
    }
}
